// SelfAttention_66116726554651
// MI455X (gfx1250) — matrix operand order repaired and checked by reading, not compiled, not run on hardware
//
#include <hip/hip_runtime.h>

typedef _Float16 v16h __attribute__((ext_vector_type(16)));
typedef _Float16 h8   __attribute__((ext_vector_type(8)));
typedef float    v8f  __attribute__((ext_vector_type(8)));

#define B_   4
#define T_   2048
#define E_   256
#define H_   8
#define D_   32
#define NROW (B_ * T_)          // 8192
#define JOINED_ 25
#define NEG_BIG (-1.0e30f)

__device__ __forceinline__ float attn_scale() { return 0.17677669529663687f; } // 1/sqrt(32)

// ---------------- gfx1250 wait helpers ------------------------------------
__device__ __forceinline__ void wait_async0() {
#if __has_builtin(__builtin_amdgcn_s_wait_asynccnt)
    __builtin_amdgcn_s_wait_asynccnt(0);
#else
    asm volatile("s_wait_asynccnt 0" ::: "memory");
#endif
}
__device__ __forceinline__ void wait_async2() {   // allow 2 newest outstanding
#if __has_builtin(__builtin_amdgcn_s_wait_asynccnt)
    __builtin_amdgcn_s_wait_asynccnt(2);
#else
    asm volatile("s_wait_asynccnt 2" ::: "memory");
#endif
}

// Async copy 16B: global -> LDS (per-lane addresses), tracked by ASYNCcnt.
__device__ __forceinline__ void async_g2l_b128(unsigned int ldsOff, const void* gptr) {
    asm volatile("global_load_async_to_lds_b128 %0, %1, off"
                 :: "v"(ldsOff), "v"(gptr) : "memory");
}

// ---- WMMA fragment loaders (f16, 16x16x32) -------------------------------
// A: 16(M) x 32(K). lane&15 = M row; kb = (lane>=16)?16:0; halves {kb..kb+7},{kb+8..kb+15}
__device__ __forceinline__ v16h frag_a_g(const _Float16* __restrict__ rowPtr, int lane) {
    const int kb = (lane >= 16) ? 16 : 0;
    h8 lo = *(const h8*)(rowPtr + kb);
    h8 hi = *(const h8*)(rowPtr + kb + 8);
    v16h a;
#pragma unroll
    for (int i = 0; i < 8; ++i) { a[i] = lo[i]; a[i + 8] = hi[i]; }
    return a;
}

// B: 32(K) x 16(N). lane&15 = N col; K = kb + e, kb = (lane>=16)?16:0.
__device__ __forceinline__ v16h frag_b_g(const _Float16* __restrict__ rowPtr, int lane) {
    const int kb = (lane >= 16) ? 16 : 0;
    return *(const v16h*)(rowPtr + kb);
}

__device__ __forceinline__ v16h frag_b_lds(const _Float16* rowPtr, int lane) {
    const int kb = (lane >= 16) ? 16 : 0;
    return *(const v16h*)(rowPtr + kb);
}

// ---- f32 -> f16 conversion -----------------------------------------------
__global__ void k_f32_to_f16(const float* __restrict__ src, _Float16* __restrict__ dst, int n) {
    int i = blockIdx.x * blockDim.x + threadIdx.x;
    if (i < n) dst[i] = (_Float16)src[i];
}

// ---- QKV projection: y = x @ W^T + b, 2x2 register-blocked (32x32/wave) ---
__device__ __forceinline__ void store_qkv(_Float16* __restrict__ dst, int i, int o,
                                          float v, int transposeOut) {
    const int bidx = i >> 11;          // / T_
    const int t    = i & (T_ - 1);
    const int h    = o >> 5;
    const int dd   = o & 31;
    size_t idx;
    if (transposeOut)
        idx = ((size_t)(bidx * H_ + h) * D_ + dd) * T_ + t;
    else
        idx = ((size_t)(bidx * H_ + h) * T_ + t) * D_ + dd;
    dst[idx] = (_Float16)v;
}

__global__ __launch_bounds__(128) void k_gemm_qkv(const _Float16* __restrict__ xh,
                                                  const _Float16* __restrict__ wh,
                                                  const float* __restrict__ bias,
                                                  _Float16* __restrict__ dst,
                                                  int transposeOut) {
    const int lane = threadIdx.x & 31;
    const int wid  = threadIdx.x >> 5;
    const int w    = blockIdx.x * 4 + wid;
    const int nt = w & 7;    // 8 col tiles of 32 (E=256)
    const int mt = w >> 3;   // 256 row tiles of 32 (8192 rows)
    const int i0 = mt * 32, o0 = nt * 32;
    const int n  = lane & 15;
    const int m  = lane & 15;
    const int hi8 = (lane >= 16) ? 8 : 0;

    const _Float16* arow0 = xh + (size_t)(i0 + m) * E_;
    const _Float16* arow1 = xh + (size_t)(i0 + 16 + m) * E_;
    const _Float16* brow0 = wh + (size_t)(o0 + n) * E_;
    const _Float16* brow1 = wh + (size_t)(o0 + 16 + n) * E_;

    v8f c00 = {}, c01 = {}, c10 = {}, c11 = {};
#pragma unroll
    for (int kk = 0; kk < E_; kk += 32) {
        v16h a0 = frag_a_g(arow0 + kk, lane);
        v16h a1 = frag_a_g(arow1 + kk, lane);
        v16h b0 = frag_b_g(brow0 + kk, lane);
        v16h b1 = frag_b_g(brow1 + kk, lane);
        c00 = __builtin_amdgcn_wmma_f32_16x16x32_f16(false, a0, false, b0, (short)0, c00, false, false);
        c01 = __builtin_amdgcn_wmma_f32_16x16x32_f16(false, a0, false, b1, (short)0, c01, false, false);
        c10 = __builtin_amdgcn_wmma_f32_16x16x32_f16(false, a1, false, b0, (short)0, c10, false, false);
        c11 = __builtin_amdgcn_wmma_f32_16x16x32_f16(false, a1, false, b1, (short)0, c11, false, false);
    }
    const float bb0 = bias[o0 + n];
    const float bb1 = bias[o0 + 16 + n];
#pragma unroll
    for (int r = 0; r < 8; ++r) {
        const int ia = i0 + r + hi8;
        const int ib = i0 + 16 + r + hi8;
        store_qkv(dst, ia, o0 + n,      c00[r] + bb0, transposeOut);
        store_qkv(dst, ia, o0 + 16 + n, c01[r] + bb1, transposeOut);
        store_qkv(dst, ib, o0 + n,      c10[r] + bb0, transposeOut);
        store_qkv(dst, ib, o0 + 16 + n, c11[r] + bb1, transposeOut);
    }
}

// ---- Flash attention -------------------------------------------------------
// Block = 4 waves, same (b,h). Wave wid owns 32 t-rows (two 16-row tiles).
// K/V 32-col step tiles are double-buffered in LDS via async copies:
// prefetch step i+1 while computing step i; each staged tile feeds 8 WMMAs.
__global__ __launch_bounds__(128) void k_attn(const _Float16* __restrict__ qh,
                                              const _Float16* __restrict__ kh,
                                              const _Float16* __restrict__ vth,
                                              _Float16* __restrict__ ah) {
    // [0,8192):  two KV buffers, each {K tile 2KB | Vt tile 2KB}
    // [8192,16384): per-wave 2KB P transpose buffers (2 row-tiles x 1KB)
    __shared__ alignas(16) unsigned char smem[16384];

    const int tid  = threadIdx.x;
    const int lane = tid & 31;
    const int wid  = tid >> 5;
    const int bh        = blockIdx.x >> 4;   // 32 (b,h) pairs
    const int blockTile = blockIdx.x & 15;   // 16 blocks of 128 t-rows
    const int base = blockTile * 128;
    const int t0a  = base + wid * 32;        // first 16-row tile
    const int t0b  = t0a + 16;               // second 16-row tile
    const int n    = lane & 15;
    const int m    = lane & 15;
    const int hi8  = (lane >= 16) ? 8 : 0;
    const int kbp  = (lane >= 16) ? 16 : 0;  // A-frag K phase for LDS reload

    const _Float16* qb = qh  + (size_t)bh * T_ * D_;
    const _Float16* kb = kh  + (size_t)bh * T_ * D_;
    const _Float16* vb = vth + (size_t)bh * D_ * T_;
    _Float16* pw = (_Float16*)(smem + 8192) + wid * 1024;   // 2KB per wave

    const unsigned int ldsBase = (unsigned int)(uintptr_t)(void*)smem;
    const int chunkRow = tid >> 2;           // 0..31 (row of 32-row tile)
    const int chunkOff = (tid & 3) * 16;     // byte offset within 64B row

    // Q fragments for the two row tiles (d==32 == one WMMA K-depth)
    v16h aq[2];
    aq[0] = frag_a_g(qb + (size_t)(t0a + m) * D_, lane);
    aq[1] = frag_a_g(qb + (size_t)(t0b + m) * D_, lane);

    v8f acc[2][2] = {};
    float mrow[2][8], lrow[2][8];
#pragma unroll
    for (int rt = 0; rt < 2; ++rt)
#pragma unroll
        for (int r = 0; r < 8; ++r) { mrow[rt][r] = NEG_BIG; lrow[rt][r] = 0.f; }

    const int s_end_blk = base + 128;        // max causal limit over the block
    const int nsteps = s_end_blk >> 5;

    // stage(s0, buf): cooperative async copy of K[32x32] + Vt[32x32] tiles
    auto stage = [&](int s0, int buf) {
        const char* gk = (const char*)kb + (size_t)(s0 + chunkRow) * (D_ * 2) + chunkOff;
        const char* gv = (const char*)vb + ((size_t)chunkRow * T_ + s0) * 2 + chunkOff;
        const unsigned int dst = ldsBase + (unsigned)(buf * 4096);
        async_g2l_b128(dst + (unsigned)(tid * 16), gk);
        async_g2l_b128(dst + 2048u + (unsigned)(tid * 16), gv);
    };

    stage(0, 0);   // prologue: fill buffer 0

    for (int it = 0; it < nsteps; ++it) {
        const int s0  = it * 32;
        const int buf = it & 1;
        const bool pre = (it + 1) < nsteps;      // uniform across block
        if (pre) {
            stage(s0 + 32, (it + 1) & 1);        // prefetch next tile
            wait_async2();                       // older pair (this tile) done
        } else {
            wait_async0();
        }
        __syncthreads();                         // all waves' copies visible

        const _Float16* Klds = (const _Float16*)(smem + buf * 4096);
        const _Float16* Vlds = Klds + 1024;      // +2048 bytes

        if (s0 < t0b + 16) {                     // any of this wave's tiles active
            // shared B-fragments: 2 K subtiles + 2 Vt d-halves -> feed 8 WMMAs
            v16h bk0 = frag_b_lds(Klds + n * 32, lane);
            v16h bk1 = frag_b_lds(Klds + (16 + n) * 32, lane);
            v16h bvf0 = frag_b_lds(Vlds + n * 32, lane);
            v16h bvf1 = frag_b_lds(Vlds + (16 + n) * 32, lane);

            const int j0 = s0 + n, j1 = s0 + 16 + n;
            const bool c0ok = (j0 % JOINED_) != (JOINED_ - 1);
            const bool c1ok = (j1 % JOINED_) != (JOINED_ - 1);

#pragma unroll
            for (int rt = 0; rt < 2; ++rt) {
                const int t0r = t0a + rt * 16;
                if (s0 < t0r + 16) {             // wave-uniform
                    v8f sc[2];
                    sc[0] = __builtin_amdgcn_wmma_f32_16x16x32_f16(false, aq[rt], false, bk0, (short)0, (v8f){}, false, false);
                    sc[1] = __builtin_amdgcn_wmma_f32_16x16x32_f16(false, aq[rt], false, bk1, (short)0, (v8f){}, false, false);

                    // ---- online softmax ----
#pragma unroll
                    for (int r = 0; r < 8; ++r) {
                        const int trow = t0r + r + hi8;
                        const bool ok0 = c0ok && (j0 <= trow);
                        const bool ok1 = c1ok && (j1 <= trow);
                        float v0 = ok0 ? sc[0][r] * attn_scale() : NEG_BIG;
                        float v1 = ok1 ? sc[1][r] * attn_scale() : NEG_BIG;
                        float lm = fmaxf(v0, v1);
#pragma unroll
                        for (int msk = 1; msk < 16; msk <<= 1)
                            lm = fmaxf(lm, __shfl_xor(lm, msk, 32));
                        const float mn = fmaxf(mrow[rt][r], lm);
                        const float alpha = __expf(mrow[rt][r] - mn);
                        const float p0 = ok0 ? __expf(v0 - mn) : 0.f;
                        const float p1 = ok1 ? __expf(v1 - mn) : 0.f;
                        float rs = p0 + p1;
#pragma unroll
                        for (int msk = 1; msk < 16; msk <<= 1)
                            rs += __shfl_xor(rs, msk, 32);
                        lrow[rt][r] = lrow[rt][r] * alpha + rs;
                        mrow[rt][r] = mn;
                        acc[rt][0][r] *= alpha;
                        acc[rt][1][r] *= alpha;
                        sc[0][r] = p0;
                        sc[1][r] = p1;
                    }

                    // ---- transpose P through per-wave LDS: C -> A layout ----
                    _Float16* pwr = pw + rt * 512;
#pragma unroll
                    for (int r = 0; r < 8; ++r) {
                        pwr[(r + hi8) * 32 + n]      = (_Float16)sc[0][r];
                        pwr[(r + hi8) * 32 + 16 + n] = (_Float16)sc[1][r];
                    }
                    asm volatile("s_wait_dscnt 0" ::: "memory");   // cross-lane RAW

                    v16h ap;
                    {
                        h8 lo = *(const h8*)(pwr + m * 32 + kbp);
                        h8 hi = *(const h8*)(pwr + m * 32 + kbp + 8);
#pragma unroll
                        for (int i = 0; i < 8; ++i) { ap[i] = lo[i]; ap[i + 8] = hi[i]; }
                    }

                    acc[rt][0] = __builtin_amdgcn_wmma_f32_16x16x32_f16(false, ap, false, bvf0, (short)0, acc[rt][0], false, false);
                    acc[rt][1] = __builtin_amdgcn_wmma_f32_16x16x32_f16(false, ap, false, bvf1, (short)0, acc[rt][1], false, false);
                }
            }
        }
        __syncthreads();   // reads done before this buffer is overwritten
    }

    // ---- normalize + store attention output as f16 [B][T][E] ----
    const int bidx = bh >> 3;    // / H
    const int h    = bh & 7;
#pragma unroll
    for (int rt = 0; rt < 2; ++rt) {
#pragma unroll
        for (int r = 0; r < 8; ++r) {
            const int t = t0a + rt * 16 + r + hi8;
            const float inv = 1.f / lrow[rt][r];
            const size_t obase = ((size_t)(bidx * T_ + t)) * E_ + h * D_;
            ah[obase + n]      = (_Float16)(acc[rt][0][r] * inv);
            ah[obase + 16 + n] = (_Float16)(acc[rt][1][r] * inv);
        }
    }
}

// ---- Output projection: out = attn @ Wo^T + bo, 2x2 blocked, f32 out ------
__global__ __launch_bounds__(128) void k_gemm_out(const _Float16* __restrict__ ahx,
                                                  const _Float16* __restrict__ woh,
                                                  const float* __restrict__ bo,
                                                  float* __restrict__ out) {
    const int lane = threadIdx.x & 31;
    const int wid  = threadIdx.x >> 5;
    const int w    = blockIdx.x * 4 + wid;
    const int nt = w & 7;
    const int mt = w >> 3;
    const int i0 = mt * 32, o0 = nt * 32;
    const int n  = lane & 15;
    const int m  = lane & 15;
    const int hi8 = (lane >= 16) ? 8 : 0;

    const _Float16* arow0 = ahx + (size_t)(i0 + m) * E_;
    const _Float16* arow1 = ahx + (size_t)(i0 + 16 + m) * E_;
    const _Float16* brow0 = woh + (size_t)(o0 + n) * E_;
    const _Float16* brow1 = woh + (size_t)(o0 + 16 + n) * E_;

    v8f c00 = {}, c01 = {}, c10 = {}, c11 = {};
#pragma unroll
    for (int kk = 0; kk < E_; kk += 32) {
        v16h a0 = frag_a_g(arow0 + kk, lane);
        v16h a1 = frag_a_g(arow1 + kk, lane);
        v16h b0 = frag_b_g(brow0 + kk, lane);
        v16h b1 = frag_b_g(brow1 + kk, lane);
        c00 = __builtin_amdgcn_wmma_f32_16x16x32_f16(false, a0, false, b0, (short)0, c00, false, false);
        c01 = __builtin_amdgcn_wmma_f32_16x16x32_f16(false, a0, false, b1, (short)0, c01, false, false);
        c10 = __builtin_amdgcn_wmma_f32_16x16x32_f16(false, a1, false, b0, (short)0, c10, false, false);
        c11 = __builtin_amdgcn_wmma_f32_16x16x32_f16(false, a1, false, b1, (short)0, c11, false, false);
    }
    const float bb0 = bo[o0 + n];
    const float bb1 = bo[o0 + 16 + n];
#pragma unroll
    for (int r = 0; r < 8; ++r) {
        const int ia = i0 + r + hi8;
        const int ib = i0 + 16 + r + hi8;
        out[(size_t)ia * E_ + (o0 + n)]      = c00[r] + bb0;
        out[(size_t)ia * E_ + (o0 + 16 + n)] = c01[r] + bb1;
        out[(size_t)ib * E_ + (o0 + n)]      = c10[r] + bb0;
        out[(size_t)ib * E_ + (o0 + 16 + n)] = c11[r] + bb1;
    }
}

// ---- host launcher --------------------------------------------------------
extern "C" void kernel_launch(void* const* d_in, const int* in_sizes, int n_in,
                              void* d_out, int out_size, void* d_ws, size_t ws_size,
                              hipStream_t stream) {
    (void)in_sizes; (void)n_in; (void)out_size; (void)ws_size;
    const float* x  = (const float*)d_in[0];
    const float* Wq = (const float*)d_in[1];
    const float* bq = (const float*)d_in[2];
    const float* Wk = (const float*)d_in[3];
    const float* bk = (const float*)d_in[4];
    const float* Wv = (const float*)d_in[5];
    const float* bv = (const float*)d_in[6];
    const float* Wo = (const float*)d_in[7];
    const float* bo = (const float*)d_in[8];
    float* out = (float*)d_out;

    const size_t NXE = (size_t)NROW * E_;   // 2,097,152
    const size_t NW  = (size_t)E_ * E_;     // 65,536

    _Float16* ws  = (_Float16*)d_ws;
    _Float16* xh  = ws;
    _Float16* wqh = xh  + NXE;
    _Float16* wkh = wqh + NW;
    _Float16* wvh = wkh + NW;
    _Float16* woh = wvh + NW;
    _Float16* qh  = woh + NW;
    _Float16* khp = qh  + NXE;
    _Float16* vth = khp + NXE;
    _Float16* ahx = vth + NXE;
    // total: 5*NXE + 4*NW halves = ~20.5 MB

    k_f32_to_f16<<<(int)((NXE + 255) / 256), 256, 0, stream>>>(x,  xh,  (int)NXE);
    k_f32_to_f16<<<(int)((NW  + 255) / 256), 256, 0, stream>>>(Wq, wqh, (int)NW);
    k_f32_to_f16<<<(int)((NW  + 255) / 256), 256, 0, stream>>>(Wk, wkh, (int)NW);
    k_f32_to_f16<<<(int)((NW  + 255) / 256), 256, 0, stream>>>(Wv, wvh, (int)NW);
    k_f32_to_f16<<<(int)((NW  + 255) / 256), 256, 0, stream>>>(Wo, woh, (int)NW);

    // 256 row-tiles x 8 col-tiles (32x32 per wave) = 2048 waves = 512 blocks
    k_gemm_qkv<<<512, 128, 0, stream>>>(xh, wqh, bq, qh,  0);
    k_gemm_qkv<<<512, 128, 0, stream>>>(xh, wkh, bk, khp, 0);
    k_gemm_qkv<<<512, 128, 0, stream>>>(xh, wvh, bv, vth, 1);

    // 32 (b,h) x 16 block-tiles (128 t-rows each) = 512 blocks of 4 waves
    k_attn<<<512, 128, 0, stream>>>(qh, khp, vth, ahx);

    k_gemm_out<<<512, 128, 0, stream>>>(ahx, woh, bo, out);
}
